// MultiHeadAttention_65257733095523
// MI455X (gfx1250) — compile-verified, matrix operand order checked by reading, not run on hardware
//
#include <hip/hip_runtime.h>
#include <hip/hip_bf16.h>

// ---------------------------------------------------------------------------
// Types for CDNA5 WMMA (wave32): v_wmma_f32_16x16x32_bf16
//   A: 16x32 bf16  (v16bf, 8 VGPRs)   lane = M row (both halves), halves split K
//   B: 32x16 bf16  (v16bf)            lane = N col (both halves), halves split K
//   C/D: 16x16 f32 (v8f)              vgpr r: M = r + 8*(lane>=16), N = lane&15
// ---------------------------------------------------------------------------
typedef __attribute__((ext_vector_type(16))) __bf16 v16bf;
typedef __attribute__((ext_vector_type(4)))  __bf16 v4bf;
typedef __attribute__((ext_vector_type(8)))  float  v8f;

__device__ __forceinline__ v8f wmma_bf16(v16bf a, v16bf b, v8f c) {
    // (neg_a, A, neg_b, B, c_mod, C, reuse_a, reuse_b)
    return __builtin_amdgcn_wmma_f32_16x16x32_bf16(false, a, false, b, (short)0, c,
                                                   false, false);
}

// Load a 16x32 bf16 fragment (A-layout, or B-layout from [N,K] row-major
// memory). base points at (row0, k0); ld = row stride in elements.
// Per-lane: row = lane&15; elements 0..7 hold K = half*8 + 0..7,
//           elements 8..15 hold K = 16 + half*8 + 0..7  -> two b128 loads.
__device__ __forceinline__ v16bf ldfrag(const __bf16* __restrict__ base, int ld) {
    const int lane = threadIdx.x & 31;
    const int row  = lane & 15;
    const int koff = (lane >> 4) << 3;   // half * 8
    const __bf16* p = base + (size_t)row * ld + koff;
    union { v16bf v; uint4 q[2]; } u;
    u.q[0] = *(const uint4*)(p);
    u.q[1] = *(const uint4*)(p + 16);
    return u.v;
}

// Prefetch the cacheline of a fragment 2 K-steps ahead (global_prefetch_b8).
__device__ __forceinline__ void pffrag(const __bf16* base, int ld) {
    const int lane = threadIdx.x & 31;
    const int row  = lane & 15;
    __builtin_prefetch(base + (size_t)row * ld, 0, 1);
}

// ---------------------------------------------------------------------------
// fp32 -> bf16 conversion (vectorized x4)
// ---------------------------------------------------------------------------
__global__ void cvt_f32_bf16(const float* __restrict__ in, __bf16* __restrict__ out,
                             int n4) {
    int i = blockIdx.x * blockDim.x + threadIdx.x;
    if (i < n4) {
        float4 f = ((const float4*)in)[i];
        v4bf o;
        o.x = (__bf16)f.x; o.y = (__bf16)f.y; o.z = (__bf16)f.z; o.w = (__bf16)f.w;
        ((v4bf*)out)[i] = o;
    }
}

// ---------------------------------------------------------------------------
// GEMM: C[M,N] = A[M,K] @ B[N,K]^T + bias[N]   (bf16 in, f32 out)
// WG = 256 threads = 8 waves arranged 2x4 (M x N); each wave owns a 64x64
// tile (4x4 fragments -> 16 WMMAs per 8 fragment loads per K-step).
// WG tile = 128x256. Grid = (M/128, N/256).
// ---------------------------------------------------------------------------
__global__ void __launch_bounds__(256)
gemm_bf16_nt(const __bf16* __restrict__ A, const __bf16* __restrict__ Bt,
             const float* __restrict__ bias, float* __restrict__ C,
             int M, int N, int K) {
    const int wave = threadIdx.x >> 5;
    const int lane = threadIdx.x & 31;
    const int wm = wave & 1, wn = wave >> 1;           // 2 x 4 wave grid
    const int m0 = blockIdx.x * 128 + wm * 64;
    const int n0 = blockIdx.y * 256 + wn * 64;

    v8f acc[4][4] = {};
    for (int kb = 0; kb < K; kb += 32) {
        v16bf af[4], bf[4];
        #pragma unroll
        for (int i = 0; i < 4; i++)
            af[i] = ldfrag(A + (size_t)(m0 + 16 * i) * K + kb, K);
        #pragma unroll
        for (int j = 0; j < 4; j++)
            bf[j] = ldfrag(Bt + (size_t)(n0 + 16 * j) * K + kb, K);
        if (kb + 64 < K) {                             // warm L2/L0 ahead
            pffrag(A  + (size_t)m0 * K + kb + 64, K);
            pffrag(Bt + (size_t)n0 * K + kb + 64, K);
        }
        #pragma unroll
        for (int i = 0; i < 4; i++)
            #pragma unroll
            for (int j = 0; j < 4; j++)
                acc[i][j] = wmma_bf16(af[i], bf[j], acc[i][j]);
    }
    const int half = lane >> 4, nl = lane & 15;
    #pragma unroll
    for (int i = 0; i < 4; i++) {
        #pragma unroll
        for (int j = 0; j < 4; j++) {
            int n = n0 + 16 * j + nl;
            float bv = bias[n];
            #pragma unroll
            for (int r = 0; r < 8; r++) {
                int m = m0 + 16 * i + r + 8 * half;
                C[(size_t)m * N + n] = acc[i][j][r] + bv;
            }
        }
    }
}

// ---------------------------------------------------------------------------
// RoPE + repack:
//   q_bh[B,H,S,64] = bf16( rope(qf) + u_bias )
//   k_bh[B,H,S,64] = bf16( rope(kf) )
//   v_t [B,H,64,S] = bf16( vf )          (transposed so P@V loads contiguously)
// One thread per (b,s,h,d); total 8*1024*16*64 = 2^23 threads.
// ---------------------------------------------------------------------------
__global__ void __launch_bounds__(256)
rope_pack(const float* __restrict__ qf, const float* __restrict__ kf,
          const float* __restrict__ vf, const float* __restrict__ u_bias,
          __bf16* __restrict__ q_bh, __bf16* __restrict__ k_bh,
          __bf16* __restrict__ v_t) {
    int gid = blockIdx.x * 256 + threadIdx.x;
    int d = gid & 63;
    int h = (gid >> 6) & 15;
    int s = (gid >> 10) & 1023;
    int b = gid >> 20;

    size_t src = ((size_t)(b * 1024 + s)) * 1024 + h * 64 + d;
    int i2 = d >> 1;                                 // rotary pair index
    float inv = __powf(10000.0f, -(float)(2 * i2) * (1.0f / 64.0f));
    float ang = (float)s * inv;
    float sn, cs;
    __sincosf(ang, &sn, &cs);

    float xq = qf[src], xk = kf[src];
    float xq2 = qf[src ^ 1], xk2 = kf[src ^ 1];      // pair partner
    float rotq = (d & 1) ? xq2 : -xq2;               // even: -x[d+1], odd: x[d-1]
    float rotk = (d & 1) ? xk2 : -xk2;
    float oq = xq * cs + rotq * sn + u_bias[h * 64 + d];
    float ok = xk * cs + rotk * sn;

    size_t bh = (size_t)(b * 16 + h);
    q_bh[(bh * 1024 + s) * 64 + d] = (__bf16)oq;
    k_bh[(bh * 1024 + s) * 64 + d] = (__bf16)ok;
    v_t[(bh * 64 + d) * 1024 + s]  = (__bf16)vf[src];
}

// ---------------------------------------------------------------------------
// Flash attention (streaming softmax). Grid = (S/64, B*H), block = 128 (4 waves).
// Each wave owns 16 query rows; loops over 16 KV tiles of 64 keys.
// Row statistics live per-lane because C-layout maps vgpr r -> M = r + 8*half,
// which is exactly the mapping produced by the 16-lane-half shfl_xor reduce.
// ---------------------------------------------------------------------------
__global__ void __launch_bounds__(128)
flash_attn(const __bf16* __restrict__ q_bh, const __bf16* __restrict__ k_bh,
           const __bf16* __restrict__ v_t, const unsigned char* __restrict__ mask,
           __bf16* __restrict__ ctx) {
    __shared__ __bf16 pbuf[4][16 * 64];              // per-wave P staging (2KB each)

    const int wave = threadIdx.x >> 5;
    const int lane = threadIdx.x & 31;
    const int half = lane >> 4, nl = lane & 15;
    const int bh = blockIdx.y;
    const int b = bh >> 4, h = bh & 15;
    const int m0 = blockIdx.x * 64 + wave * 16;      // query row base of this wave

    const __bf16* qbase = q_bh + ((size_t)bh * 1024 + m0) * 64;
    v16bf qa0 = ldfrag(qbase +  0, 64);
    v16bf qa1 = ldfrag(qbase + 32, 64);

    v8f ctxa0 = {}, ctxa1 = {}, ctxa2 = {}, ctxa3 = {};
    v8f* ctxa[4] = {&ctxa0, &ctxa1, &ctxa2, &ctxa3};

    float rmax[8], rsum[8];
    #pragma unroll
    for (int r = 0; r < 8; r++) { rmax[r] = -__builtin_inff(); rsum[r] = 0.0f; }

    const unsigned char* mrow = mask + (size_t)b * 1024 * 1024;

    for (int t = 0; t < 1024; t += 64) {
        // ---- S = Q @ K^T for this 16x64 tile (8 WMMAs) ----
        v8f sf[4];
        #pragma unroll
        for (int j = 0; j < 4; j++) {
            const __bf16* kb = k_bh + ((size_t)bh * 1024 + t + j * 16) * 64;
            v16bf b0 = ldfrag(kb +  0, 64);
            v16bf b1 = ldfrag(kb + 32, 64);
            v8f z = {};
            z = wmma_bf16(qa0, b0, z);
            sf[j] = wmma_bf16(qa1, b1, z);
        }
        // ---- mask + scale (1/sqrt(64) = 0.125) ----
        #pragma unroll
        for (int j = 0; j < 4; j++) {
            int n = t + j * 16 + nl;
            #pragma unroll
            for (int r = 0; r < 8; r++) {
                int m = m0 + r + 8 * half;
                unsigned char mk = mrow[(size_t)m * 1024 + n];
                sf[j][r] = mk ? -1.0e9f : sf[j][r] * 0.125f;
            }
        }
        // ---- online softmax stats ----
        float nmax[8], corr[8], tsum[8];
        #pragma unroll
        for (int r = 0; r < 8; r++) {
            float v = fmaxf(fmaxf(sf[0][r], sf[1][r]), fmaxf(sf[2][r], sf[3][r]));
            v = fmaxf(v, __shfl_xor(v, 1, 32));
            v = fmaxf(v, __shfl_xor(v, 2, 32));
            v = fmaxf(v, __shfl_xor(v, 4, 32));
            v = fmaxf(v, __shfl_xor(v, 8, 32));      // stays within 16-lane half
            nmax[r] = fmaxf(rmax[r], v);
            corr[r] = __expf(rmax[r] - nmax[r]);
            rmax[r] = nmax[r];
            tsum[r] = 0.0f;
        }
        #pragma unroll
        for (int j = 0; j < 4; j++) {
            #pragma unroll
            for (int r = 0; r < 8; r++) {
                float p = __expf(sf[j][r] - nmax[r]);
                sf[j][r] = p;
                tsum[r] += p;
            }
        }
        #pragma unroll
        for (int r = 0; r < 8; r++) {
            float v = tsum[r];
            v += __shfl_xor(v, 1, 32);
            v += __shfl_xor(v, 2, 32);
            v += __shfl_xor(v, 4, 32);
            v += __shfl_xor(v, 8, 32);
            rsum[r] = rsum[r] * corr[r] + v;
        }
        // ---- rescale accumulated context ----
        #pragma unroll
        for (int nf = 0; nf < 4; nf++)
            #pragma unroll
            for (int r = 0; r < 8; r++)
                (*ctxa[nf])[r] *= corr[r];
        // ---- stage P (bf16) through LDS to convert C-layout -> A-layout ----
        __bf16* pl = &pbuf[wave][0];
        #pragma unroll
        for (int j = 0; j < 4; j++) {
            #pragma unroll
            for (int r = 0; r < 8; r++) {
                int m = r + 8 * half;
                pl[m * 64 + j * 16 + nl] = (__bf16)sf[j][r];
            }
        }
        __syncthreads();
        v16bf pa0 = ldfrag(pl +  0, 64);
        v16bf pa1 = ldfrag(pl + 32, 64);
        // ---- ctx += P @ V  (V^T is [Dh, S] row-major -> contiguous loads) ----
        #pragma unroll
        for (int nf = 0; nf < 4; nf++) {
            const __bf16* vb = v_t + ((size_t)bh * 64 + nf * 16) * 1024 + t;
            v16bf v0 = ldfrag(vb +  0, 1024);
            v16bf v1 = ldfrag(vb + 32, 1024);
            *ctxa[nf] = wmma_bf16(pa0, v0, *ctxa[nf]);
            *ctxa[nf] = wmma_bf16(pa1, v1, *ctxa[nf]);
        }
        __syncthreads();
    }
    // ---- normalize and write ctx as [B,S,H*Dh] bf16 ----
    #pragma unroll
    for (int nf = 0; nf < 4; nf++) {
        int n = h * 64 + nf * 16 + nl;
        #pragma unroll
        for (int r = 0; r < 8; r++) {
            int m = m0 + r + 8 * half;
            float o = (*ctxa[nf])[r] / rsum[r];
            ctx[((size_t)b * 1024 + m) * 1024 + n] = (__bf16)o;
        }
    }
}

// ---------------------------------------------------------------------------
// Launch: converts -> 3 projection GEMMs -> rope/pack -> flash attn -> out GEMM
// Workspace layout (bytes), peak 152 MB with aliasing:
//   [0,16M)    qbf   (query bf16)     -> reused as q_bh
//   [16,32M)   kbf                     -> reused as k_bh
//   [32,48M)   vbf                     -> reused as v_t
//   [48,50M)   Wq bf16   [50,52M) Wk   [52,54M) Wv   [54,56M) Wo
//   [56,88M)   qf f32 proj            -> reused as ctx bf16 (16M)
//   [88,120M)  kf f32 proj
//   [120,152M) vf f32 proj
// ---------------------------------------------------------------------------
extern "C" void kernel_launch(void* const* d_in, const int* in_sizes, int n_in,
                              void* d_out, int out_size, void* d_ws, size_t ws_size,
                              hipStream_t stream) {
    const float* query  = (const float*)d_in[0];
    const float* key    = (const float*)d_in[1];
    const float* value  = (const float*)d_in[2];
    const unsigned char* mask = (const unsigned char*)d_in[3];
    const float* bq = (const float*)d_in[5];
    const float* bk = (const float*)d_in[7];
    const float* bv = (const float*)d_in[9];
    const float* u_bias = (const float*)d_in[10];
    const float* bo = (const float*)d_in[12];
    const float* Wq = (const float*)d_in[4];
    const float* Wk = (const float*)d_in[6];
    const float* Wv = (const float*)d_in[8];
    const float* Wo = (const float*)d_in[11];

    const size_t MB = 1u << 20;
    char* ws = (char*)d_ws;
    __bf16* qbf  = (__bf16*)(ws + 0 * MB);
    __bf16* kbf  = (__bf16*)(ws + 16 * MB);
    __bf16* vbf  = (__bf16*)(ws + 32 * MB);
    __bf16* wqb  = (__bf16*)(ws + 48 * MB);
    __bf16* wkb  = (__bf16*)(ws + 50 * MB);
    __bf16* wvb  = (__bf16*)(ws + 52 * MB);
    __bf16* wob  = (__bf16*)(ws + 54 * MB);
    float*  qf   = (float*) (ws + 56 * MB);
    float*  kf   = (float*) (ws + 88 * MB);
    float*  vf   = (float*) (ws + 120 * MB);
    __bf16* q_bh = qbf;                       // alias (safe: sequenced)
    __bf16* k_bh = kbf;
    __bf16* v_t  = vbf;
    __bf16* ctxb = (__bf16*)(ws + 56 * MB);   // alias qf (dead after rope_pack)

    const int M = 8192, N = 1024, K = 1024;   // B*S, D, D
    const int actN4 = (8 * 1024 * 1024) / 4;  // activation elems / 4
    const int wN4   = (1024 * 1024) / 4;      // weight elems / 4

    cvt_f32_bf16<<<(actN4 + 255) / 256, 256, 0, stream>>>(query, qbf, actN4);
    cvt_f32_bf16<<<(actN4 + 255) / 256, 256, 0, stream>>>(key,   kbf, actN4);
    cvt_f32_bf16<<<(actN4 + 255) / 256, 256, 0, stream>>>(value, vbf, actN4);
    cvt_f32_bf16<<<(wN4 + 255) / 256, 256, 0, stream>>>(Wq, wqb, wN4);
    cvt_f32_bf16<<<(wN4 + 255) / 256, 256, 0, stream>>>(Wk, wkb, wN4);
    cvt_f32_bf16<<<(wN4 + 255) / 256, 256, 0, stream>>>(Wv, wvb, wN4);
    cvt_f32_bf16<<<(wN4 + 255) / 256, 256, 0, stream>>>(Wo, wob, wN4);

    dim3 ggrid(M / 128, N / 256);
    gemm_bf16_nt<<<ggrid, 256, 0, stream>>>(qbf, wqb, bq, qf, M, N, K);
    gemm_bf16_nt<<<ggrid, 256, 0, stream>>>(kbf, wkb, bk, kf, M, N, K);
    gemm_bf16_nt<<<ggrid, 256, 0, stream>>>(vbf, wvb, bv, vf, M, N, K);

    rope_pack<<<(8 * 1024 * 16 * 64) / 256, 256, 0, stream>>>(
        qf, kf, vf, u_bias, q_bh, k_bh, v_t);

    flash_attn<<<dim3(1024 / 64, 8 * 16), 128, 0, stream>>>(
        q_bh, k_bh, v_t, mask, ctxb);

    gemm_bf16_nt<<<ggrid, 256, 0, stream>>>(ctxb, wob, bo, (float*)d_out, M, N, K);
}
